// MultiHeadAttention_73366631350643
// MI455X (gfx1250) — matrix operand order repaired and checked by reading, not compiled, not run on hardware
//
#include <hip/hip_runtime.h>

// ---------------------------------------------------------------------------
// MHA forward for MI455X (gfx1250, wave32, WMMA 16x16x32 f16 -> f32 accum)
// B=2, S=2048, D_MODEL=1024, H=16, D_K=64
// Round 3: fix async builtin pointer types (int4 AS1 / AS3).
// ---------------------------------------------------------------------------

typedef __attribute__((ext_vector_type(4)))  _Float16 v4h;
typedef __attribute__((ext_vector_type(8)))  _Float16 v8h;
typedef __attribute__((ext_vector_type(16))) _Float16 v16h;
typedef __attribute__((ext_vector_type(8)))  float    v8f;
typedef __attribute__((ext_vector_type(4)))  int      v4i;

#define SEQ     2048
#define DMODEL  1024
#define NHEAD   16
#define DK      64
#define MROWS   4096          // B * SEQ
#define LDSP    48            // padded LDS row stride (halves); 96B rows, 16B aligned chunks

#if __has_builtin(__builtin_amdgcn_global_load_async_to_lds_b128) && \
    __has_builtin(__builtin_amdgcn_s_wait_asynccnt)
#define HAVE_ASYNC 1
#else
#define HAVE_ASYNC 0
#endif

static __device__ __forceinline__ void async_copy_b128(void* lds, const void* g) {
#if HAVE_ASYNC
    __builtin_amdgcn_global_load_async_to_lds_b128(
        (__attribute__((address_space(1))) v4i*)g,
        (__attribute__((address_space(3))) v4i*)lds, 0, 0);
#else
    *(v8h*)lds = *(const v8h*)g;
#endif
}
static __device__ __forceinline__ void wait_async_all() {
#if HAVE_ASYNC
    __builtin_amdgcn_s_wait_asynccnt(0);
#endif
}

static __device__ __forceinline__ v16h frag_combine(v8h lo, v8h hi) {
    v16h r;
#pragma unroll
    for (int i = 0; i < 8; ++i) { r[i] = lo[i]; r[i + 8] = hi[i]; }
    return r;
}
static __device__ __forceinline__ v16h load_frag(const _Float16* p0, const _Float16* p1) {
    return frag_combine(*(const v8h*)p0, *(const v8h*)p1);
}
static __device__ __forceinline__ float redmax16(float v) {
    v = fmaxf(v, __shfl_xor(v, 1, 32));
    v = fmaxf(v, __shfl_xor(v, 2, 32));
    v = fmaxf(v, __shfl_xor(v, 4, 32));
    v = fmaxf(v, __shfl_xor(v, 8, 32));
    return v;
}
static __device__ __forceinline__ float redsum16(float v) {
    v += __shfl_xor(v, 1, 32);
    v += __shfl_xor(v, 2, 32);
    v += __shfl_xor(v, 4, 32);
    v += __shfl_xor(v, 8, 32);
    return v;
}

// ---------------------------------------------------------------------------
// f32 -> f16 conversion pass (weights), float4 vectorized.
// ---------------------------------------------------------------------------
__global__ __launch_bounds__(256) void cvt_f32_f16(const float* __restrict__ src,
                                                   _Float16* __restrict__ dst) {
    const size_t i = (size_t)blockIdx.x * blockDim.x + threadIdx.x;
    float4 v = *(const float4*)(src + i * 4);
    v4h h = { (_Float16)v.x, (_Float16)v.y, (_Float16)v.z, (_Float16)v.w };
    *(v4h*)(dst + i * 4) = h;
}

// ---------------------------------------------------------------------------
// GEMM: out = A(4096x1024) * W^T(1024x1024) + bias.  W pre-converted to f16.
// MODE 0: out f16 [b, h, s, dk]   (Q and K)
// MODE 2: out f16 [b, h, dk, s]   (V transposed)
// MODE 3: out f32 [b, s, d_model] (final projection)
// AHALF:  A operand is f16 (attention output) -> async-staged like W.
// Tile: 128(M) x 128(N), K-step 32, 8 waves, double-buffered LDS.
// ---------------------------------------------------------------------------
template <int MODE, bool AHALF>
__global__ __launch_bounds__(256) void gemm_wmma(const float* __restrict__ Af,
                                                 const _Float16* __restrict__ Ah,
                                                 const _Float16* __restrict__ Wh,
                                                 const float* __restrict__ bias,
                                                 void* __restrict__ out) {
    __shared__ _Float16 sA[2][128 * LDSP];
    __shared__ _Float16 sB[2][128 * LDSP];

    const int tid  = threadIdx.x;
    const int wv   = tid >> 5;
    const int lane = tid & 31;
    const int hi   = lane >> 4;     // which 16-lane half
    const int lr   = lane & 15;
    const int m0   = blockIdx.x * 128;
    const int n0   = blockIdx.y * 128;

    v8f acc[8] = {};

    // tile stager: fills sA[buf]/sB[buf] for K-offset kk
    auto stage = [&](int kk, int buf) {
        // ---- A tile (128 x 32) ----
        if (AHALF) {
#pragma unroll
            for (int it = 0; it < 2; ++it) {
                int i   = tid + it * 256;       // 512 chunks of 8 halves
                int row = i >> 2;
                int c8  = i & 3;
                async_copy_b128(&sA[buf][row * LDSP + c8 * 8],
                                Ah + (size_t)(m0 + row) * DMODEL + kk + c8 * 8);
            }
        } else {
#pragma unroll
            for (int it = 0; it < 4; ++it) {
                int i   = tid + it * 256;       // 1024 float4 chunks
                int row = i >> 3;
                int c4  = i & 7;
                float4 v = *(const float4*)(Af + (size_t)(m0 + row) * DMODEL + kk + c4 * 4);
                v4h hv = { (_Float16)v.x, (_Float16)v.y, (_Float16)v.z, (_Float16)v.w };
                *(v4h*)(&sA[buf][row * LDSP + c4 * 4]) = hv;
            }
        }
        // ---- W tile (128 N-rows x 32 K), f16, async ----
#pragma unroll
        for (int it = 0; it < 2; ++it) {
            int i   = tid + it * 256;
            int row = i >> 2;
            int c8  = i & 3;
            async_copy_b128(&sB[buf][row * LDSP + c8 * 8],
                            Wh + (size_t)(n0 + row) * DMODEL + kk + c8 * 8);
        }
    };

    stage(0, 0);
    wait_async_all();
    __syncthreads();

    for (int k0 = 0; k0 < DMODEL; k0 += 32) {
        const int cur = (k0 >> 5) & 1;
        if (k0 + 32 < DMODEL) stage(k0 + 32, cur ^ 1);

        // A fragment: 16x32, lane layout per CDNA5 ISA 7.12.2 (contiguous 8-half chunks)
        const _Float16* ap = &sA[cur][(wv * 16 + lr) * LDSP + hi * 8];
        v16h a = load_frag(ap, ap + 16);
#pragma unroll
        for (int nt = 0; nt < 8; ++nt) {
            // B fragment: 32x16 K-major; lane n holds 8-half K chunks hi*8 and 16+hi*8 (matches A)
            const _Float16* bp = &sB[cur][(nt * 16 + lr) * LDSP + hi * 8];
            v16h b = load_frag(bp, bp + 16);
            acc[nt] = __builtin_amdgcn_wmma_f32_16x16x32_f16(
                false, a, false, b, (short)0, acc[nt], false, false);
        }
        wait_async_all();
        __syncthreads();
    }

    // ---- epilogue: bias + mode-dependent scatter ----
#pragma unroll
    for (int nt = 0; nt < 8; ++nt) {
        const int n  = n0 + nt * 16 + lr;       // global column
        const float bv = bias[n];
        const int hh = n >> 6;                  // head
        const int dk = n & 63;
#pragma unroll
        for (int r = 0; r < 8; ++r) {
            const int gm = m0 + wv * 16 + r + 8 * hi;   // global row (C layout)
            const int b  = gm >> 11;
            const int s  = gm & (SEQ - 1);
            const float val = acc[nt][r] + bv;
            if (MODE == 0) {
                ((_Float16*)out)[(((size_t)b * NHEAD + hh) * SEQ + s) * DK + dk] = (_Float16)val;
            } else if (MODE == 2) {
                ((_Float16*)out)[(((size_t)b * NHEAD + hh) * DK + dk) * SEQ + s] = (_Float16)val;
            } else {
                ((float*)out)[(size_t)gm * DMODEL + n] = val;
            }
        }
    }
}

// ---------------------------------------------------------------------------
// Flash attention: one wave per 16-query tile. Online softmax, P*V via WMMA.
// q,k: [b,h,s,dk] f16 ; vt: [b,h,dk,s] f16 ; attn out: [b,s,h*dk] f16
// ---------------------------------------------------------------------------
__global__ __launch_bounds__(128) void attn_wmma(const _Float16* __restrict__ q,
                                                 const _Float16* __restrict__ k,
                                                 const _Float16* __restrict__ vt,
                                                 const int* __restrict__ mask,
                                                 _Float16* __restrict__ attn) {
    __shared__ _Float16 sp[4][16 * LDSP];       // per-wave P tile (16x32)

    const int tid  = threadIdx.x;
    const int wv   = tid >> 5;
    const int lane = tid & 31;
    const int hi   = lane >> 4;
    const int lr   = lane & 15;
    const int bh   = blockIdx.x >> 5;           // (b*H + h)
    const int qc   = blockIdx.x & 31;
    const int q0   = (qc * 4 + wv) * 16;
    const int b    = bh >> 4;
    const int h    = bh & 15;

    const _Float16* qp  = q  + (size_t)bh * SEQ * DK;
    const _Float16* kp  = k  + (size_t)bh * SEQ * DK;
    const _Float16* vtp = vt + (size_t)bh * DK * SEQ;
    const int*      mp  = mask + (size_t)b * SEQ * SEQ;

    // Q fragments (16 rows x 64 d -> two 16x32 A fragments), prescaled by 1/8
    const _Float16* qa = qp + (size_t)(q0 + lr) * DK;
    v16h aQ0 = load_frag(qa + hi * 8,      qa + hi * 8 + 16);
    v16h aQ1 = load_frag(qa + 32 + hi * 8, qa + 32 + hi * 8 + 16);
#pragma unroll
    for (int i = 0; i < 16; ++i) { aQ0[i] = aQ0[i] * (_Float16)0.125f; aQ1[i] = aQ1[i] * (_Float16)0.125f; }

    float run_max[8], run_sum[8];
#pragma unroll
    for (int r = 0; r < 8; ++r) { run_max[r] = -1e30f; run_sum[r] = 0.0f; }
    v8f oacc[4] = {};

    for (int kt = 0; kt < SEQ; kt += 32) {
        // ---- scores: two 16x16 tiles over 32 key columns ----
        v8f sc0 = {}, sc1 = {};
        {
            const _Float16* kb0 = kp + (size_t)(kt + lr) * DK + hi * 8;
            v16h b0 = load_frag(kb0,      kb0 + 16);
            v16h b1 = load_frag(kb0 + 32, kb0 + 48);
            sc0 = __builtin_amdgcn_wmma_f32_16x16x32_f16(false, aQ0, false, b0, (short)0, sc0, false, false);
            sc0 = __builtin_amdgcn_wmma_f32_16x16x32_f16(false, aQ1, false, b1, (short)0, sc0, false, false);
            const _Float16* kb1 = kp + (size_t)(kt + 16 + lr) * DK + hi * 8;
            v16h b2 = load_frag(kb1,      kb1 + 16);
            v16h b3 = load_frag(kb1 + 32, kb1 + 48);
            sc1 = __builtin_amdgcn_wmma_f32_16x16x32_f16(false, aQ0, false, b2, (short)0, sc1, false, false);
            sc1 = __builtin_amdgcn_wmma_f32_16x16x32_f16(false, aQ1, false, b3, (short)0, sc1, false, false);
        }
        // ---- mask ----
#pragma unroll
        for (int r = 0; r < 8; ++r) {
            const int* mr = mp + (size_t)(q0 + r + 8 * hi) * SEQ + kt + lr;
            if (mr[0]  == 0) sc0[r] = -1e9f;
            if (mr[16] == 0) sc1[r] = -1e9f;
        }
        // ---- online softmax over this 32-column block ----
        float corr[8];
#pragma unroll
        for (int r = 0; r < 8; ++r) {
            float tmax = redmax16(fmaxf(sc0[r], sc1[r]));
            float mn   = fmaxf(run_max[r], tmax);
            corr[r]    = __expf(run_max[r] - mn);
            run_max[r] = mn;
            float p0 = __expf(sc0[r] - mn);
            float p1 = __expf(sc1[r] - mn);
            sc0[r] = p0; sc1[r] = p1;
            run_sum[r] = run_sum[r] * corr[r] + redsum16(p0 + p1);
        }
#pragma unroll
        for (int nt = 0; nt < 4; ++nt)
#pragma unroll
            for (int r = 0; r < 8; ++r) oacc[nt][r] *= corr[r];

        // ---- reshape P (C layout) -> A fragment via per-wave LDS tile ----
        _Float16* pw = &sp[wv][0];
#pragma unroll
        for (int r = 0; r < 8; ++r) {
            const int row = r + 8 * hi;
            pw[row * LDSP + lr]      = (_Float16)sc0[r];
            pw[row * LDSP + 16 + lr] = (_Float16)sc1[r];
        }
        __syncthreads();
        const _Float16* pr = &sp[wv][lr * LDSP + hi * 8];
        v16h aP = load_frag(pr, pr + 16);

        // ---- O += P * V (4 tiles of 16 dk columns; K dim = 32 key rows) ----
#pragma unroll
        for (int nt = 0; nt < 4; ++nt) {
            const _Float16* vb = vtp + (size_t)(nt * 16 + lr) * SEQ + kt + hi * 8;
            v16h bv = load_frag(vb, vb + 16);
            oacc[nt] = __builtin_amdgcn_wmma_f32_16x16x32_f16(
                false, aP, false, bv, (short)0, oacc[nt], false, false);
        }
        __syncthreads();
    }

    // ---- normalize and write [b, s, h*64 + dk] f16 ----
#pragma unroll
    for (int r = 0; r < 8; ++r) {
        const float inv = 1.0f / run_sum[r];
        const int row = q0 + r + 8 * hi;
        _Float16* op = attn + ((size_t)(b * SEQ + row)) * DMODEL + h * DK;
#pragma unroll
        for (int nt = 0; nt < 4; ++nt)
            op[nt * 16 + lr] = (_Float16)(oacc[nt][r] * inv);
    }
}

// ---------------------------------------------------------------------------
extern "C" void kernel_launch(void* const* d_in, const int* in_sizes, int n_in,
                              void* d_out, int out_size, void* d_ws, size_t ws_size,
                              hipStream_t stream) {
    (void)in_sizes; (void)n_in; (void)out_size; (void)ws_size;
    const float* query = (const float*)d_in[0];
    const float* key   = (const float*)d_in[1];
    const float* value = (const float*)d_in[2];
    const int*   mask  = (const int*)  d_in[3];
    const float* w_q   = (const float*)d_in[4];
    const float* b_q   = (const float*)d_in[5];
    const float* w_k   = (const float*)d_in[6];
    const float* b_k   = (const float*)d_in[7];
    const float* w_v   = (const float*)d_in[8];
    const float* b_v   = (const float*)d_in[9];
    const float* w_o   = (const float*)d_in[10];
    const float* b_o   = (const float*)d_in[11];

    _Float16* ws = (_Float16*)d_ws;
    const size_t SEG  = (size_t)MROWS * DMODEL;    // 4M halves = 8MB
    const size_t WSEG = (size_t)DMODEL * DMODEL;   // 1M halves = 2MB
    _Float16* q_ws = ws;
    _Float16* k_ws = ws + SEG;
    _Float16* vT   = ws + 2 * SEG;
    _Float16* attn = ws + 3 * SEG;
    _Float16* whq  = ws + 4 * SEG;
    _Float16* whk  = whq + WSEG;
    _Float16* whv  = whk + WSEG;
    _Float16* who  = whv + WSEG;

    // weight f32 -> f16 pre-conversion (1M elements each, 1024 blocks x 256 thr x 4)
    const int cvtg = (DMODEL * DMODEL / 4) / 256;
    cvt_f32_f16<<<cvtg, 256, 0, stream>>>(w_q, whq);
    cvt_f32_f16<<<cvtg, 256, 0, stream>>>(w_k, whk);
    cvt_f32_f16<<<cvtg, 256, 0, stream>>>(w_v, whv);
    cvt_f32_f16<<<cvtg, 256, 0, stream>>>(w_o, who);

    dim3 gg(MROWS / 128, DMODEL / 128);            // (32, 8)

    gemm_wmma<0, false><<<gg, 256, 0, stream>>>(query, nullptr, whq, b_q, (void*)q_ws);
    gemm_wmma<0, false><<<gg, 256, 0, stream>>>(key,   nullptr, whk, b_k, (void*)k_ws);
    gemm_wmma<2, false><<<gg, 256, 0, stream>>>(value, nullptr, whv, b_v, (void*)vT);

    attn_wmma<<<dim3(2 * NHEAD * (SEQ / 64)), 128, 0, stream>>>(q_ws, k_ws, vT, mask, attn);

    gemm_wmma<3, true><<<gg, 256, 0, stream>>>(nullptr, attn, who, b_o, d_out);
}
